// InnerModel_58815282152048
// MI455X (gfx1250) — matrix operand order repaired and checked by reading, not compiled, not run on hardware
//
#include <hip/hip_runtime.h>

// -------------------------------------------------------------------------
// MI455X (gfx1250) implementation of the PCC instance-aggregation layer.
//
// Phase 0 (prep):  convert P_company / P_person / W_PCC to fp16, pre-swizzled
//                  into the 16x16x32 WMMA B-fragment lane layout.
// Phase 1 (proj):  f_c = taxF @ P_c, f_p = perF @ P_p via
//                  v_wmma_f32_16x16x32_f16; fp16 tables (25.6 MB each) live
//                  in workspace and stay resident in the 192 MB L2.
// Phase 2 (edges): per 16-edge tile, gather 3 fp16 rows/edge from the
//                  L2-resident tables, 24 WMMAs vs LDS-cached W fragments,
//                  LeakyReLU, nontemporal f32 stores.
//
// Roofline: ~550 MB of HBM traffic -> ~24 us @ 23.3 TB/s.  The 768 MB of
// random gathers hit L2 only — protected by NT hints on the big streaming
// loads/stores so they cannot evict the f-tables.
//
// All tile-level control values are forced into SGPRs via readfirstlane so
// tile loops / tail handling compile to scalar branches (no exec juggling).
// -------------------------------------------------------------------------

typedef __attribute__((ext_vector_type(16))) _Float16 v16h;
typedef __attribute__((ext_vector_type(8)))  _Float16 v8h;
typedef __attribute__((ext_vector_type(8)))  float    v8f;
typedef __attribute__((ext_vector_type(4)))  float    v4f;

#define IN_DIM 128
#define D      64
#define ALPHA  0.2f

// Fragment geometry (wave32, 16x16x32 f16 WMMA):
//   B frag: lane l holds column n = nt*16 + (l&15); element j holds
//           K = kbase + ((l>>4)<<3) + (j&7) + ((j>>3)<<4).   (contiguous 16 halfs per lane)
//   A frag: lane l holds row m = l&15; elements 0..7 -> K = kbase+(l>>4)*8+j,
//           elements 8..15 -> K = kbase+(l>>4)*8+16+j.
//   C/D:    lane l holds column n = l&15; element r holds row m = r+(l>>4)*8.

__device__ __forceinline__ v8f wmma16(v16h a, v16h b, v8f c) {
  return __builtin_amdgcn_wmma_f32_16x16x32_f16(false, a, false, b, (short)0,
                                                c, false, false);
}

// ---------------- Phase 0: pack weights into fp16 B-fragments --------------
__global__ __launch_bounds__(256) void prep_frags(
    const float* __restrict__ Pc, const float* __restrict__ Pp,
    const float* __restrict__ W,  _Float16* __restrict__ pcf,
    _Float16* __restrict__ ppf,   _Float16* __restrict__ wf) {
  int idx = blockIdx.x * blockDim.x + threadIdx.x;
  if (idx >= 28672) return;
  if (idx < 16384) {                    // P_company / P_person regions
    int region = idx >> 13;             // 0 = Pc, 1 = Pp
    int local  = idx & 8191;
    int f = local >> 9, l = (local >> 4) & 31, j = local & 15;
    int kc = f >> 2, nt = f & 3;
    int k = kc * 32 + ((l >> 4) << 3) + (j & 7) + ((j >> 3) << 4);
    int n = nt * 16 + (l & 15);
    const float* src = region ? Pp : Pc;
    _Float16*    dst = region ? ppf : pcf;
    dst[local] = (_Float16)src[k * D + n];
  } else {                              // W_PCC region (192 x 64)
    int local = idx - 16384;            // 0..12287
    int f = local >> 9, l = (local >> 4) & 31, j = local & 15;
    int s = f >> 3, kc = (f >> 2) & 1, nt = f & 3;
    int k = s * 64 + kc * 32 + ((l >> 4) << 3) + (j & 7) + ((j >> 3) << 4);
    int n = nt * 16 + (l & 15);
    wf[local] = (_Float16)W[k * D + n];
  }
}

// ---------------- Phase 1: feature projection GEMMs ------------------------
// One wave per 16-row output tile.  [M=16, K=128, N=64] per tile.
__global__ __launch_bounds__(256) void proj_gemm(
    const float* __restrict__ featC, const float* __restrict__ featP,
    const _Float16* __restrict__ pcf, const _Float16* __restrict__ ppf,
    _Float16* __restrict__ fc, _Float16* __restrict__ fp,
    int NC, int NP, int tilesC, int tilesTot) {
  // Scalarize all wave-level control state.
  int wave = __builtin_amdgcn_readfirstlane(
      (int)(blockIdx.x * (blockDim.x >> 5) + (threadIdx.x >> 5)));
  if (wave >= tilesTot) return;                 // scalar branch
  int lane = threadIdx.x & 31;
  int mat  = (wave >= tilesC) ? 1 : 0;          // scalar
  int t    = mat ? (wave - tilesC) : wave;      // scalar
  const float*    feats = mat ? featP : featC;
  const _Float16* frg   = mat ? ppf : pcf;
  _Float16*       tbl   = mat ? fp : fc;
  int N = mat ? NP : NC;                        // scalar

  int rowbase = t * 16;                         // scalar
  bool full = (rowbase + 16 <= N);              // scalar tail flag
  int row = rowbase + (lane & 15);
  if (row >= N) row = N - 1;                    // clamp keeps EXEC full
  int kb = (lane >> 4) << 3;                    // 0 or 8
  const float* rp = feats + (size_t)row * IN_DIM;

  // A fragments: streamed-once f32 features -> NT loads (don't pollute L2,
  // which must keep the fp16 f-tables for phase 2).
  v16h a[4];
#pragma unroll
  for (int kc = 0; kc < 4; ++kc) {
    int ko = kc * 32 + kb;
    v4f f0 = __builtin_nontemporal_load((const v4f*)(rp + ko));
    v4f f1 = __builtin_nontemporal_load((const v4f*)(rp + ko + 4));
    v4f f2 = __builtin_nontemporal_load((const v4f*)(rp + ko + 16));
    v4f f3 = __builtin_nontemporal_load((const v4f*)(rp + ko + 20));
    v16h aa;
#pragma unroll
    for (int j = 0; j < 4; ++j) {
      aa[j]      = (_Float16)f0[j];
      aa[4 + j]  = (_Float16)f1[j];
      aa[8 + j]  = (_Float16)f2[j];
      aa[12 + j] = (_Float16)f3[j];
    }
    a[kc] = aa;
  }

#pragma unroll
  for (int nt = 0; nt < 4; ++nt) {
    // Load all 4 B fragments into distinct registers first (one clause, one
    // wait), then issue the 4 WMMAs back-to-back.
    v16h b0 = *(const v16h*)(frg + ((0 * 4 + nt) << 9) + lane * 16);
    v16h b1 = *(const v16h*)(frg + ((1 * 4 + nt) << 9) + lane * 16);
    v16h b2 = *(const v16h*)(frg + ((2 * 4 + nt) << 9) + lane * 16);
    v16h b3 = *(const v16h*)(frg + ((3 * 4 + nt) << 9) + lane * 16);
    v8f acc = {};
    acc = wmma16(a[0], b0, acc);
    acc = wmma16(a[1], b1, acc);
    acc = wmma16(a[2], b2, acc);
    acc = wmma16(a[3], b3, acc);

    int n = nt * 16 + (lane & 15);
    // f-table stores stay default-temporal: we WANT them resident in L2.
    if (full) {                                 // scalar branch
#pragma unroll
      for (int r = 0; r < 8; ++r)
        tbl[(size_t)(rowbase + r + kb) * D + n] = (_Float16)acc[r];
    } else {
#pragma unroll
      for (int r = 0; r < 8; ++r) {
        int orow = rowbase + r + kb;
        if (orow < N) tbl[(size_t)orow * D + n] = (_Float16)acc[r];
      }
    }
  }
}

// ---------------- Phase 2: gathered edge GEMM + LeakyReLU ------------------
// One wave per 16-edge tile (grid-stride).  W fragments live in LDS.
__global__ __launch_bounds__(256) void edge_agg(
    const _Float16* __restrict__ fc, const _Float16* __restrict__ fp,
    const _Float16* __restrict__ wf,
    const int* __restrict__ i0, const int* __restrict__ i1,
    const int* __restrict__ ip, float* __restrict__ out,
    int E, int ntiles) {
  __shared__ _Float16 wlds[24 * 512];           // 24 KB of W fragments
  {
    const uint4* src = (const uint4*)wf;
    uint4*       dst = (uint4*)wlds;
    for (int i = threadIdx.x; i < (24 * 512 * 2) / 16; i += blockDim.x)
      dst[i] = src[i];
  }
  __syncthreads();

  int lane = threadIdx.x & 31;
  // Scalarize the grid-stride loop controls.
  int wave = __builtin_amdgcn_readfirstlane(
      (int)(blockIdx.x * (blockDim.x >> 5) + (threadIdx.x >> 5)));
  int stride = __builtin_amdgcn_readfirstlane(
      (int)(gridDim.x * (blockDim.x >> 5)));
  int m16 = lane & 15;
  int kb  = (lane >> 4) << 3;                   // 0 or 8

  for (int t = wave; t < ntiles; t += stride) { // scalar loop
    int e0 = t * 16;                            // scalar
    bool full = (e0 + 16 <= E);                 // scalar tail flag
    int eL = e0 + m16;
    if (eL >= E) eL = E - 1;                    // clamp keeps EXEC full

    // Prefetch next tile's indices (global_prefetch_b8) to hide the
    // idx -> gather dependent-load chain.
    if (t + stride < ntiles) {                  // scalar branch
      int eN = (t + stride) * 16 + m16;
      __builtin_prefetch(i0 + eN, 0, 1);
      __builtin_prefetch(i1 + eN, 0, 1);
      __builtin_prefetch(ip + eN, 0, 1);
    }

    size_t r0 = (size_t)__builtin_nontemporal_load(i0 + eL);
    size_t r1 = (size_t)__builtin_nontemporal_load(i1 + eL);
    size_t r2 = (size_t)__builtin_nontemporal_load(ip + eL);
    const _Float16* p0 = fc + r0 * D;
    const _Float16* p1 = fc + r1 * D;
    const _Float16* p2 = fp + r2 * D;

    // Gathered A fragments: default temporal policy -> served from L2.
    v16h a[3][2];
#pragma unroll
    for (int kc = 0; kc < 2; ++kc) {
      int ko = kc * 32 + kb;
      v8h lo0 = *(const v8h*)(p0 + ko), hi0 = *(const v8h*)(p0 + ko + 16);
      v8h lo1 = *(const v8h*)(p1 + ko), hi1 = *(const v8h*)(p1 + ko + 16);
      v8h lo2 = *(const v8h*)(p2 + ko), hi2 = *(const v8h*)(p2 + ko + 16);
      v16h a0, a1, a2;
#pragma unroll
      for (int j = 0; j < 8; ++j) {
        a0[j] = lo0[j]; a0[j + 8] = hi0[j];
        a1[j] = lo1[j]; a1[j + 8] = hi1[j];
        a2[j] = lo2[j]; a2[j + 8] = hi2[j];
      }
      a[0][kc] = a0; a[1][kc] = a1; a[2][kc] = a2;
    }

#pragma unroll
    for (int nt = 0; nt < 4; ++nt) {
      // All 6 LDS B fragments up front, then 6 WMMAs back-to-back.
      const _Float16* wl = wlds + (nt << 9) + lane * 16;
      v16h b00 = *(const v16h*)(wl + (0 << 11));
      v16h b01 = *(const v16h*)(wl + (1 << 11));
      v16h b10 = *(const v16h*)(wl + (2 << 11));
      v16h b11 = *(const v16h*)(wl + (3 << 11));
      v16h b20 = *(const v16h*)(wl + (4 << 11));
      v16h b21 = *(const v16h*)(wl + (5 << 11));
      v8f acc = {};
      acc = wmma16(a[0][0], b00, acc);
      acc = wmma16(a[0][1], b01, acc);
      acc = wmma16(a[1][0], b10, acc);
      acc = wmma16(a[1][1], b11, acc);
      acc = wmma16(a[2][0], b20, acc);
      acc = wmma16(a[2][1], b21, acc);

      int n = nt * 16 + m16;
      // Output is write-once (256 MB): NT stores so it never evicts the
      // L2-resident gather tables.
      if (full) {                               // scalar branch
#pragma unroll
        for (int r = 0; r < 8; ++r) {
          float v = acc[r];
          v = (v >= 0.f) ? v : ALPHA * v;       // LeakyReLU
          __builtin_nontemporal_store(v, out + (size_t)(e0 + r + kb) * D + n);
        }
      } else {
#pragma unroll
        for (int r = 0; r < 8; ++r) {
          int e = e0 + r + kb;
          float v = acc[r];
          v = (v >= 0.f) ? v : ALPHA * v;
          if (e < E)
            __builtin_nontemporal_store(v, out + (size_t)e * D + n);
        }
      }
    }
  }
}

// ---------------------------- host launcher --------------------------------
extern "C" void kernel_launch(void* const* d_in, const int* in_sizes, int n_in,
                              void* d_out, int out_size, void* d_ws,
                              size_t ws_size, hipStream_t stream) {
  const float* taxF = (const float*)d_in[0];
  const float* perF = (const float*)d_in[1];
  // d_in[2] (item_feats) and d_in[8] (P_item) are dead for the PCC output.
  const int* i0 = (const int*)d_in[3];
  const int* i1 = (const int*)d_in[4];
  const int* ip = (const int*)d_in[5];
  const float* Pc = (const float*)d_in[6];
  const float* Pp = (const float*)d_in[7];
  const float* W  = (const float*)d_in[9];
  float* out = (float*)d_out;

  int NC = in_sizes[0] / IN_DIM;
  int NP = in_sizes[1] / IN_DIM;
  int E  = in_sizes[3];

  // Workspace layout (halfs): [fc | fp | pc_frags(8192) | pp_frags(8192) |
  //                            w_frags(12288)]  -> ~51.3 MB total.
  _Float16* fc  = (_Float16*)d_ws;
  _Float16* fp  = fc + (size_t)NC * D;
  _Float16* pcf = fp + (size_t)NP * D;
  _Float16* ppf = pcf + 8192;
  _Float16* wf  = ppf + 8192;

  prep_frags<<<(28672 + 255) / 256, 256, 0, stream>>>(Pc, Pp, W, pcf, ppf, wf);

  int tilesC = (NC + 15) / 16;
  int tilesP = (NP + 15) / 16;
  int tilesTot = tilesC + tilesP;
  proj_gemm<<<(tilesTot + 7) / 8, 256, 0, stream>>>(
      taxF, perF, pcf, ppf, fc, fp, NC, NP, tilesC, tilesTot);

  int ntiles = (E + 15) / 16;
  int blocks = (ntiles + 7) / 8;
  if (blocks > 2048) blocks = 2048;
  if (blocks < 1) blocks = 1;
  edge_agg<<<blocks, 256, 0, stream>>>(fc, fp, wf, i0, i1, ip, out, E, ntiles);
}
